// StackedLSTMSeq2Seq_3899830304989
// MI455X (gfx1250) — matrix operand order repaired and checked by reading, not compiled, not run on hardware
//
#include <hip/hip_runtime.h>

// ---------------------------------------------------------------------------
// MI455X (gfx1250) seq2seq LSTM.
// Precision strategy: weights/activations bf16 (converted once, L2-resident),
// accumulation f32 via v_wmma_f32_16x16x32_bf16. LSTM phase is L2-bandwidth
// bound (weights re-read per timestep), projection is HBM/L2 streaming bound.
// ---------------------------------------------------------------------------

typedef __bf16 bf16_t;
typedef bf16_t v16bf __attribute__((ext_vector_type(16)));
typedef float  v8f   __attribute__((ext_vector_type(8)));

struct alignas(16) U4 { unsigned int x, y, z, w; };
struct alignas(32) Frag32B { U4 lo, hi; };

__device__ __forceinline__ unsigned short f2bf(float f) {
    unsigned int u = __float_as_uint(f);
    unsigned int r = (u + 0x7FFFu + ((u >> 16) & 1u)) >> 16;   // RNE
    return (unsigned short)r;
}

// A fragment (16x32 bf16, M x K): lane = M + 16*c ; elems 0-7 -> K = kb+8c..,
// elems 8-15 -> K = kb+16+8c..   (ISA 7.12.2 table, 16-bit A)
__device__ __forceinline__ v16bf load_a_frag(const unsigned short* __restrict__ A,
                                             int row, int K, int kb, int lane) {
    int c = (lane >> 4) & 1;
    const unsigned short* p = A + (size_t)row * K + kb;
    Frag32B t;
    t.lo = *(const U4*)(p + 8 * c);
    t.hi = *(const U4*)(p + 16 + 8 * c);
    return __builtin_bit_cast(v16bf, t);
}

// B fragment (32x16 bf16, K x N) built from row-major W[N,K] (B = W-block^T):
// lane = N + 16*c ; elems 0-7 -> K = kb+8c.., elems 8-15 -> K = kb+16+8c.. (matches A)
__device__ __forceinline__ v16bf load_b_frag(const unsigned short* __restrict__ W,
                                             int ntb, int K, int kb, int lane) {
    const unsigned short* p = W + (size_t)(ntb + (lane & 15)) * K + kb;
    int c = (lane >> 4) & 1;
    Frag32B t;
    t.lo = *(const U4*)(p + 8 * c);
    t.hi = *(const U4*)(p + 16 + 8 * c);
    return __builtin_bit_cast(v16bf, t);
}

__device__ __forceinline__ v8f wmma_bf16(v16bf a, v16bf b, v8f c) {
    return __builtin_amdgcn_wmma_f32_16x16x32_bf16(false, a, false, b, (short)0, c, false, false);
}

// ---------------------------------------------------------------------------
// Preprocessing kernels
// ---------------------------------------------------------------------------

// dst = bf16(a [+ b])   (weights; b==nullptr for plain conversion)
__global__ void k_cvt_bf16_add(const float* __restrict__ a, const float* __restrict__ b,
                               unsigned short* __restrict__ dst, long n) {
    long i = (long)blockIdx.x * blockDim.x + threadIdx.x;
    if (i >= n) return;
    float v = a[i];
    if (b) v += b[i];
    dst[i] = f2bf(v);
}

__global__ void k_add_f32(const float* __restrict__ a, const float* __restrict__ b,
                          float* __restrict__ dst, int n) {
    int i = blockIdx.x * blockDim.x + threadIdx.x;
    if (i < n) dst[i] = a[i] + b[i];
}

// dst[(t*B + b)*E + e] = bf16(emb[tok[b*S + t]*E + e])
__global__ void k_gather_embed(const int* __restrict__ tok, const float* __restrict__ emb,
                               unsigned short* __restrict__ dst, int S, int B, int E) {
    long i = (long)blockIdx.x * blockDim.x + threadIdx.x;
    long total = (long)S * B * E;
    if (i >= total) return;
    int e  = (int)(i % E);
    int tb = (int)(i / E);
    int b  = tb % B;
    int t  = tb / B;
    dst[i] = f2bf(emb[(size_t)tok[b * S + t] * E + e]);
}

__global__ void k_init_state(const float* __restrict__ h0, const float* __restrict__ c0,
                             unsigned short* __restrict__ hbf, float* __restrict__ c, int n) {
    int i = blockIdx.x * blockDim.x + threadIdx.x;
    if (i >= n) return;
    hbf[i] = f2bf(h0[i]);
    c[i]   = c0[i];
}

// ---------------------------------------------------------------------------
// LSTM gate GEMM:  G[32,4096] = Xbf[32,1024] @ Wih^T  (+ Hbf[32,1024] @ Whh^T)
// One wave per 16-wide N tile (256 waves), two M tiles (rows 0-15 / 16-31).
// ---------------------------------------------------------------------------
__global__ void k_lstm_gates(const unsigned short* __restrict__ X,
                             const unsigned short* __restrict__ H,
                             const unsigned short* __restrict__ Wih,
                             const unsigned short* __restrict__ Whh,
                             float* __restrict__ G) {
    const int N = 4096, K = 1024;
    int wave = (int)((blockIdx.x * blockDim.x + threadIdx.x) >> 5);
    int lane = threadIdx.x & 31;
    if (wave >= (N >> 4)) return;          // wave-uniform
    int ntb = wave << 4;
    int r0  = lane & 15;
    int c   = (lane >> 4) & 1;

    v8f acc0 = {0, 0, 0, 0, 0, 0, 0, 0};
    v8f acc1 = {0, 0, 0, 0, 0, 0, 0, 0};

    for (int k = 0; k < K; k += 32) {
        v16bf bf = load_b_frag(Wih, ntb, K, k, lane);
        v16bf a0 = load_a_frag(X, r0,      K, k, lane);
        v16bf a1 = load_a_frag(X, 16 + r0, K, k, lane);
        acc0 = wmma_bf16(a0, bf, acc0);
        acc1 = wmma_bf16(a1, bf, acc1);
    }
    if (Whh) {
        for (int k = 0; k < K; k += 32) {
            v16bf bf = load_b_frag(Whh, ntb, K, k, lane);
            v16bf a0 = load_a_frag(H, r0,      K, k, lane);
            v16bf a1 = load_a_frag(H, 16 + r0, K, k, lane);
            acc0 = wmma_bf16(a0, bf, acc0);
            acc1 = wmma_bf16(a1, bf, acc1);
        }
    }
    int n = ntb + r0;
#pragma unroll
    for (int r = 0; r < 8; ++r) {
        int m = 8 * c + r;
        G[(size_t)m * N + n]        = acc0[r];
        G[(size_t)(16 + m) * N + n] = acc1[r];
    }
}

// ---------------------------------------------------------------------------
// LSTM pointwise: gates(i,f,g,o) + combined bias -> c,h ; h stored as bf16.
// Optionally scatter h into hs[b][t][:] for the decoder.
// ---------------------------------------------------------------------------
__global__ void k_lstm_pointwise(const float* __restrict__ G, const float* __restrict__ bsum,
                                 float* __restrict__ c, unsigned short* __restrict__ hbf,
                                 unsigned short* __restrict__ hs, int hs_stride) {
    int idx = blockIdx.x * blockDim.x + threadIdx.x;
    if (idx >= 32 * 1024) return;
    int b = idx >> 10, n = idx & 1023;
    const float* g = G + (size_t)b * 4096;
    float gi = g[n]        + bsum[n];
    float gf = g[n + 1024] + bsum[n + 1024];
    float gg = g[n + 2048] + bsum[n + 2048];
    float go = g[n + 3072] + bsum[n + 3072];
    float si = 1.f / (1.f + expf(-gi));
    float sf = 1.f / (1.f + expf(-gf));
    float so = 1.f / (1.f + expf(-go));
    float cn = sf * c[idx] + si * tanhf(gg);
    c[idx] = cn;
    float h = so * tanhf(cn);
    hbf[idx] = f2bf(h);
    if (hs) hs[(size_t)b * hs_stride + n] = f2bf(h);
}

// ---------------------------------------------------------------------------
// Output projection: out[1536,32000] = relu(hs_bf[1536,1024] @ outW^T + out_b)
// 4 M-tiles per wave (out_W fragment reused 4x), 2000 N-tiles x 24 M-groups.
// ---------------------------------------------------------------------------
__global__ void k_proj(const unsigned short* __restrict__ A,   // [1536,1024] bf16
                       const unsigned short* __restrict__ W,   // [32000,1024] bf16
                       const float* __restrict__ bias,
                       float* __restrict__ out) {
    const int K = 1024, V = 32000;
    int wave = (int)((blockIdx.x * blockDim.x + threadIdx.x) >> 5);
    int lane = threadIdx.x & 31;
    int nt = wave % 2000;
    int mg = wave / 2000;
    if (mg >= 24) return;                  // wave-uniform
    int ntb = nt << 4;
    int mb  = mg * 64;
    int r0  = lane & 15;
    int c   = (lane >> 4) & 1;

    v8f zero = {0, 0, 0, 0, 0, 0, 0, 0};
    v8f acc[4] = {zero, zero, zero, zero};

    for (int k = 0; k < K; k += 32) {
        v16bf bf = load_b_frag(W, ntb, K, k, lane);
#pragma unroll
        for (int j = 0; j < 4; ++j) {
            v16bf af = load_a_frag(A, mb + j * 16 + r0, K, k, lane);
            acc[j] = wmma_bf16(af, bf, acc[j]);
        }
    }
    int n = ntb + r0;
    float bn = bias[n];
#pragma unroll
    for (int j = 0; j < 4; ++j) {
#pragma unroll
        for (int r = 0; r < 8; ++r) {
            int m = mb + j * 16 + 8 * c + r;
            float v = acc[j][r] + bn;
            out[(size_t)m * V + n] = v > 0.f ? v : 0.f;
        }
    }
}

// ---------------------------------------------------------------------------
// log_softmax over V (axis=2): one block per row of 32000.
// ---------------------------------------------------------------------------
__global__ void k_logsoftmax_rows(float* __restrict__ out, int V) {
    __shared__ float red[256];
    size_t base = (size_t)blockIdx.x * V;
    int tid = threadIdx.x;

    float mx = -3.4e38f;
    for (int i = tid; i < V; i += 256) mx = fmaxf(mx, out[base + i]);
    red[tid] = mx; __syncthreads();
    for (int s = 128; s > 0; s >>= 1) {
        if (tid < s) red[tid] = fmaxf(red[tid], red[tid + s]);
        __syncthreads();
    }
    mx = red[0]; __syncthreads();

    float sum = 0.f;
    for (int i = tid; i < V; i += 256) sum += expf(out[base + i] - mx);
    red[tid] = sum; __syncthreads();
    for (int s = 128; s > 0; s >>= 1) {
        if (tid < s) red[tid] += red[tid + s];
        __syncthreads();
    }
    float lse = mx + logf(red[0]);
    for (int i = tid; i < V; i += 256) out[base + i] -= lse;
}

// log_softmax over batch axis (axis=0, 32 entries, stride = 48*32000)
__global__ void k_logsoftmax_axis0(float* __restrict__ out, int cols) {
    int col = blockIdx.x * blockDim.x + threadIdx.x;
    if (col >= cols) return;
    float v[32];
    float mx = -3.4e38f;
#pragma unroll
    for (int b = 0; b < 32; ++b) {
        v[b] = out[(size_t)b * cols + col];
        mx = fmaxf(mx, v[b]);
    }
    float s = 0.f;
#pragma unroll
    for (int b = 0; b < 32; ++b) s += expf(v[b] - mx);
    float lse = mx + logf(s);
#pragma unroll
    for (int b = 0; b < 32; ++b) out[(size_t)b * cols + col] = v[b] - lse;
}

// ---------------------------------------------------------------------------
// Host orchestration
// ---------------------------------------------------------------------------
extern "C" void kernel_launch(void* const* d_in, const int* in_sizes, int n_in,
                              void* d_out, int out_size, void* d_ws, size_t ws_size,
                              hipStream_t stream) {
    (void)in_sizes; (void)n_in; (void)out_size; (void)ws_size;

    const int*   inp     = (const int*)d_in[0];
    const int*   tar     = (const int*)d_in[1];
    const float* h0      = (const float*)d_in[2];
    const float* c0      = (const float*)d_in[3];
    const float* enc_emb = (const float*)d_in[4];
    const float* dec_emb = (const float*)d_in[5];
    const float* W_ih[4] = {(const float*)d_in[6],  (const float*)d_in[10],
                            (const float*)d_in[14], (const float*)d_in[18]};
    const float* W_hh[4] = {(const float*)d_in[7],  (const float*)d_in[11],
                            (const float*)d_in[15], (const float*)d_in[19]};
    const float* b_ih[4] = {(const float*)d_in[8],  (const float*)d_in[12],
                            (const float*)d_in[16], (const float*)d_in[20]};
    const float* b_hh[4] = {(const float*)d_in[9],  (const float*)d_in[13],
                            (const float*)d_in[17], (const float*)d_in[21]};
    const float* out_W   = (const float*)d_in[22];
    const float* out_b   = (const float*)d_in[23];
    float*       out     = (float*)d_out;

    const int B = 32, S = 48, E = 1024, H = 1024, V = 32000;
    const long WMAT = (long)4 * H * E;        // 4,194,304 elems
    const long OWN  = (long)V * H;            // 32,768,000 elems

    size_t off = 0;
    auto carve = [&](size_t bytes) {
        void* p = (char*)d_ws + off;
        off += (bytes + 255) & ~(size_t)255;
        return p;
    };
    unsigned short* Wih1e  = (unsigned short*)carve(WMAT * 2);
    unsigned short* Whh1e  = (unsigned short*)carve(WMAT * 2);
    unsigned short* Wsum2e = (unsigned short*)carve(WMAT * 2);
    unsigned short* Wih1d  = (unsigned short*)carve(WMAT * 2);
    unsigned short* Whh1d  = (unsigned short*)carve(WMAT * 2);
    unsigned short* Wsum2d = (unsigned short*)carve(WMAT * 2);
    unsigned short* outWbf = (unsigned short*)carve(OWN * 2);
    unsigned short* x_src  = (unsigned short*)carve((size_t)S * B * E * 2);
    unsigned short* x_tgt  = (unsigned short*)carve((size_t)S * B * E * 2);
    unsigned short* hs_bf  = (unsigned short*)carve((size_t)B * S * H * 2);
    unsigned short* h_bf   = (unsigned short*)carve((size_t)B * H * 2);
    float*          c_st   = (float*)carve((size_t)B * H * 4);
    float*          G      = (float*)carve((size_t)B * 4 * H * 4);
    float*          bsum[4];
    for (int i = 0; i < 4; ++i) bsum[i] = (float*)carve(4 * H * 4);

    // --- preprocessing ---
    int cvtB = (int)((WMAT + 255) / 256);
    k_cvt_bf16_add<<<cvtB, 256, 0, stream>>>(W_ih[0], nullptr, Wih1e, WMAT);
    k_cvt_bf16_add<<<cvtB, 256, 0, stream>>>(W_hh[0], nullptr, Whh1e, WMAT);
    k_cvt_bf16_add<<<cvtB, 256, 0, stream>>>(W_ih[1], W_hh[1], Wsum2e, WMAT);
    k_cvt_bf16_add<<<cvtB, 256, 0, stream>>>(W_ih[2], nullptr, Wih1d, WMAT);
    k_cvt_bf16_add<<<cvtB, 256, 0, stream>>>(W_hh[2], nullptr, Whh1d, WMAT);
    k_cvt_bf16_add<<<cvtB, 256, 0, stream>>>(W_ih[3], W_hh[3], Wsum2d, WMAT);
    k_cvt_bf16_add<<<(int)((OWN + 255) / 256), 256, 0, stream>>>(out_W, nullptr, outWbf, OWN);
    for (int i = 0; i < 4; ++i)
        k_add_f32<<<16, 256, 0, stream>>>(b_ih[i], b_hh[i], bsum[i], 4 * H);

    long gatherN = (long)S * B * E;
    int  gatherB = (int)((gatherN + 255) / 256);
    k_gather_embed<<<gatherB, 256, 0, stream>>>(inp, enc_emb, x_src, S, B, E);
    k_gather_embed<<<gatherB, 256, 0, stream>>>(tar, dec_emb, x_tgt, S, B, E);
    k_init_state<<<(B * H + 255) / 256, 256, 0, stream>>>(h0, c0, h_bf, c_st, B * H);

    // --- encoder: 48 steps, cell1 (x,h) + cell2 (h,h -> Wsum) ---
    for (int t = 0; t < S; ++t) {
        const unsigned short* xt = x_src + (size_t)t * B * E;
        k_lstm_gates<<<32, 256, 0, stream>>>(xt, h_bf, Wih1e, Whh1e, G);
        k_lstm_pointwise<<<128, 256, 0, stream>>>(G, bsum[0], c_st, h_bf, nullptr, 0);
        k_lstm_gates<<<32, 256, 0, stream>>>(h_bf, h_bf, Wsum2e, nullptr, G);
        k_lstm_pointwise<<<128, 256, 0, stream>>>(G, bsum[1], c_st, h_bf, nullptr, 0);
    }

    // --- decoder: 48 steps; collect h (after cell2) into hs_bf[b][t][:] ---
    for (int t = 0; t < S; ++t) {
        const unsigned short* xt = x_tgt + (size_t)t * B * E;
        k_lstm_gates<<<32, 256, 0, stream>>>(xt, h_bf, Wih1d, Whh1d, G);
        k_lstm_pointwise<<<128, 256, 0, stream>>>(G, bsum[2], c_st, h_bf, nullptr, 0);
        k_lstm_gates<<<32, 256, 0, stream>>>(h_bf, h_bf, Wsum2d, nullptr, G);
        k_lstm_pointwise<<<128, 256, 0, stream>>>(G, bsum[3], c_st, h_bf,
                                                  hs_bf + (size_t)t * H, S * H);
    }

    // --- projection + relu -> d_out [B*T, V] ---
    // 2000 N-tiles x 24 M-groups = 48000 waves = 6000 blocks x 8 waves
    k_proj<<<6000, 256, 0, stream>>>(hs_bf, outWbf, out_b, out);

    // --- log_softmax over V per (b,t) row, then over batch axis ---
    k_logsoftmax_rows<<<B * S, 256, 0, stream>>>(out, V);
    int cols = S * V;
    k_logsoftmax_axis0<<<(cols + 255) / 256, 256, 0, stream>>>(out, cols);
}
